// MultiHeadAttentionBlock_76089640616369
// MI455X (gfx1250) — matrix operand order repaired and checked by reading, not compiled, not run on hardware
//
#include <hip/hip_runtime.h>
#include <hip/hip_bf16.h>

typedef __attribute__((ext_vector_type(16))) __bf16 v16bf;
typedef __attribute__((ext_vector_type(8)))  float  v8f;
typedef __attribute__((ext_vector_type(4)))  unsigned int u32x4;

union FragBF {
  v16bf v;
  u32x4 q[2];
  unsigned short h[16];
};

__device__ __forceinline__ unsigned short f2bf(float f) {
  __bf16 b = (__bf16)f;                       // hardware f32->bf16 convert
  return __builtin_bit_cast(unsigned short, b);
}

constexpr int kDM = 1024;   // d_model
constexpr int kS  = 2048;   // sequence length
constexpr int TM  = 128;    // GEMM rows per block (8 waves x 16)
constexpr int TN  = 64;     // GEMM cols per block
constexpr int TK  = 64;     // GEMM K-step per staging round

// ---------------------------------------------------------------------------
// GEMM: Y[r,o] = sum_k X[r,k] * W[o,k] + bias[o]   (X f32, W f32, Y bf16)
// grid: (rows/128, 1024/64), block: 256 threads (8 waves), 128x64 C tile
// ---------------------------------------------------------------------------
__global__ __launch_bounds__(256)
void gemm_f32_bf16(const float* __restrict__ X, const float* __restrict__ W,
                   const float* __restrict__ bias, unsigned short* __restrict__ Y) {
  __shared__ alignas(16) unsigned short xs[TM][TK];  // 16 KB
  __shared__ alignas(16) unsigned short wl[TN][TK];  //  8 KB
  const int t    = threadIdx.x;
  const int lane = t & 31;
  const int wave = t >> 5;
  const int half = lane >> 4;
  const int l16  = lane & 15;
  const int rb   = blockIdx.x;
  const int cb   = blockIdx.y;

  v8f acc[4];
  for (int nt = 0; nt < 4; ++nt)
    for (int r = 0; r < 8; ++r) acc[nt][r] = 0.f;

  const int xr = t >> 1, xseg = t & 1;   // xs: 128 rows, 2 x 32-half segments
  const int wr = t >> 2, wseg = t & 3;   // wl: 64 rows, 4 x 16-half segments
  const float* xrow = X + (size_t)(rb * TM + xr) * kDM + xseg * 32;
  const float* wrow = W + (size_t)(cb * TN + wr) * kDM + wseg * 16;

  for (int k0 = 0; k0 < kDM; k0 += TK) {
    if (k0 + TK < kDM) {
      __builtin_prefetch(xrow + k0 + TK, 0, 0);
      __builtin_prefetch(wrow + k0 + TK, 0, 0);
    }
    {
      const float4* px = (const float4*)(xrow + k0);
      unsigned short* dx = &xs[xr][xseg * 32];
      for (int i = 0; i < 8; ++i) {
        float4 a = px[i];
        dx[4*i+0] = f2bf(a.x); dx[4*i+1] = f2bf(a.y);
        dx[4*i+2] = f2bf(a.z); dx[4*i+3] = f2bf(a.w);
      }
      const float4* pw = (const float4*)(wrow + k0);
      unsigned short* dw = &wl[wr][wseg * 16];
      for (int i = 0; i < 4; ++i) {
        float4 b = pw[i];
        dw[4*i+0] = f2bf(b.x); dw[4*i+1] = f2bf(b.y);
        dw[4*i+2] = f2bf(b.z); dw[4*i+3] = f2bf(b.w);
      }
    }
    __syncthreads();
    for (int kk = 0; kk < 2; ++kk) {
      FragBF a;
      a.q[0] = *(const u32x4*)&xs[wave * 16 + l16][kk * 32 + half * 8];
      a.q[1] = *(const u32x4*)&xs[wave * 16 + l16][kk * 32 + 16 + half * 8];
      for (int nt = 0; nt < 4; ++nt) {
        FragBF b;
        b.q[0] = *(const u32x4*)&wl[nt * 16 + l16][kk * 32 + half * 8];
        b.q[1] = *(const u32x4*)&wl[nt * 16 + l16][kk * 32 + 16 + half * 8];
        acc[nt] = __builtin_amdgcn_wmma_f32_16x16x32_bf16(
            false, a.v, false, b.v, (short)0, acc[nt], false, false);
      }
    }
    __syncthreads();
  }

  for (int nt = 0; nt < 4; ++nt) {
    int col = cb * TN + nt * 16 + l16;
    float bv = bias[col];
    for (int r = 0; r < 8; ++r) {
      int row = rb * TM + wave * 16 + r + 8 * half;
      Y[(size_t)row * kDM + col] = f2bf(acc[nt][r] + bv);
    }
  }
}

// ---------------------------------------------------------------------------
// GEMM: Y[r,o] = sum_k X[r,k] * W[o,k] + bias[o]   (X bf16, W f32, Y f32)
// ---------------------------------------------------------------------------
__global__ __launch_bounds__(256)
void gemm_bf16_f32(const unsigned short* __restrict__ X, const float* __restrict__ W,
                   const float* __restrict__ bias, float* __restrict__ Y) {
  __shared__ alignas(16) unsigned short xs[TM][TK];
  __shared__ alignas(16) unsigned short wl[TN][TK];
  const int t    = threadIdx.x;
  const int lane = t & 31;
  const int wave = t >> 5;
  const int half = lane >> 4;
  const int l16  = lane & 15;
  const int rb   = blockIdx.x;
  const int cb   = blockIdx.y;

  v8f acc[4];
  for (int nt = 0; nt < 4; ++nt)
    for (int r = 0; r < 8; ++r) acc[nt][r] = 0.f;

  const int xr = t >> 1, xseg = t & 1;
  const int wr = t >> 2, wseg = t & 3;
  const unsigned short* xrow = X + (size_t)(rb * TM + xr) * kDM + xseg * 32;
  const float* wrow = W + (size_t)(cb * TN + wr) * kDM + wseg * 16;

  for (int k0 = 0; k0 < kDM; k0 += TK) {
    if (k0 + TK < kDM) {
      __builtin_prefetch(xrow + k0 + TK, 0, 0);
      __builtin_prefetch(wrow + k0 + TK, 0, 0);
    }
    {
      const u32x4* px = (const u32x4*)(xrow + k0);
      u32x4* dxv = (u32x4*)&xs[xr][xseg * 32];
      for (int i = 0; i < 4; ++i) dxv[i] = px[i];
      const float4* pw = (const float4*)(wrow + k0);
      unsigned short* dw = &wl[wr][wseg * 16];
      for (int i = 0; i < 4; ++i) {
        float4 b = pw[i];
        dw[4*i+0] = f2bf(b.x); dw[4*i+1] = f2bf(b.y);
        dw[4*i+2] = f2bf(b.z); dw[4*i+3] = f2bf(b.w);
      }
    }
    __syncthreads();
    for (int kk = 0; kk < 2; ++kk) {
      FragBF a;
      a.q[0] = *(const u32x4*)&xs[wave * 16 + l16][kk * 32 + half * 8];
      a.q[1] = *(const u32x4*)&xs[wave * 16 + l16][kk * 32 + 16 + half * 8];
      for (int nt = 0; nt < 4; ++nt) {
        FragBF b;
        b.q[0] = *(const u32x4*)&wl[nt * 16 + l16][kk * 32 + half * 8];
        b.q[1] = *(const u32x4*)&wl[nt * 16 + l16][kk * 32 + 16 + half * 8];
        acc[nt] = __builtin_amdgcn_wmma_f32_16x16x32_bf16(
            false, a.v, false, b.v, (short)0, acc[nt], false, false);
      }
    }
    __syncthreads();
  }

  for (int nt = 0; nt < 4; ++nt) {
    int col = cb * TN + nt * 16 + l16;
    float bv = bias[col];
    for (int r = 0; r < 8; ++r) {
      int row = rb * TM + wave * 16 + r + 8 * half;
      Y[(size_t)row * kDM + col] = acc[nt][r] + bv;
    }
  }
}

// ---------------------------------------------------------------------------
// Attention: one workgroup per (qt, h, b); 128 threads (4 waves).
// scores(16 x 2048) in LDS -> mask -> softmax -> P @ V -> bf16 out tile.
// ---------------------------------------------------------------------------
__global__ __launch_bounds__(128)
void attn_kernel(const unsigned short* __restrict__ Qp,
                 const unsigned short* __restrict__ Kp,
                 const unsigned short* __restrict__ Vp,
                 const int* __restrict__ mask,
                 unsigned short* __restrict__ Op) {
  __shared__ float sc[16][kS];                         // 128 KB
  __shared__ alignas(16) unsigned short qt_s[16][64];  // 2 KB
  __shared__ alignas(16) unsigned short vt[4][64][32]; // 16 KB (per-wave V^T)
  __shared__ float outw[4][16][64];                    // 16 KB
  __shared__ float red[16][8];
  __shared__ float rowmax[16];
  __shared__ float rowinv[16];

  const int t    = threadIdx.x;
  const int lane = t & 31;
  const int wave = t >> 5;
  const int half = lane >> 4;
  const int l16  = lane & 15;
  const int qt   = blockIdx.x;   // 0..127
  const int h    = blockIdx.y;   // 0..15
  const int b    = blockIdx.z;   // 0..3

  const size_t rowbase = (size_t)b * kS;
  const int q0 = qt * 16;

  // stage Q tile (16 x 64 bf16)
  {
    int row = t >> 3, part = t & 7;
    const u32x4* src =
        (const u32x4*)(Qp + (rowbase + q0 + row) * kDM + h * 64 + part * 8);
    *(u32x4*)&qt_s[row][part * 8] = src[0];
  }
  __syncthreads();

  FragBF qa[2];
  for (int s = 0; s < 2; ++s) {
    qa[s].q[0] = *(const u32x4*)&qt_s[l16][s * 32 + half * 8];
    qa[s].q[1] = *(const u32x4*)&qt_s[l16][s * 32 + 16 + half * 8];
  }

  // ---- scores = Q K^T * scale, masked ----
  const float scale = 0.125f;  // 1/sqrt(64)
  for (int kt = wave * 32; kt < wave * 32 + 32; ++kt) {
    v8f acc;
    for (int r = 0; r < 8; ++r) acc[r] = 0.f;
    int key = kt * 16 + l16;
    const unsigned short* krow = Kp + (rowbase + key) * kDM + h * 64;
    for (int s = 0; s < 2; ++s) {
      FragBF kb;
      kb.q[0] = *(const u32x4*)(krow + s * 32 + half * 8);
      kb.q[1] = *(const u32x4*)(krow + s * 32 + 16 + half * 8);
      acc = __builtin_amdgcn_wmma_f32_16x16x32_bf16(
          false, qa[s].v, false, kb.v, (short)0, acc, false, false);
    }
    for (int r = 0; r < 8; ++r) {
      int qrow = r + 8 * half;
      int kcol = kt * 16 + l16;
      int m = mask[(size_t)(q0 + qrow) * kS + kcol];
      sc[qrow][kcol] = (m == 0) ? -1e9f : acc[r] * scale;
    }
  }
  __syncthreads();

  // ---- softmax over each of the 16 rows (8 threads per row) ----
  {
    int row = t >> 3, sub = t & 7;
    float mx = -3.4e38f;
    for (int j = sub; j < kS; j += 8) mx = fmaxf(mx, sc[row][j]);
    red[row][sub] = mx;
    __syncthreads();
    if (sub == 0) {
      float m2 = red[row][0];
      for (int i = 1; i < 8; ++i) m2 = fmaxf(m2, red[row][i]);
      rowmax[row] = m2;
    }
    __syncthreads();
    float m2 = rowmax[row];
    float sum = 0.f;
    for (int j = sub; j < kS; j += 8) {
      float e = __expf(sc[row][j] - m2);
      sc[row][j] = e;
      sum += e;
    }
    red[row][sub] = sum;
    __syncthreads();
    if (sub == 0) {
      float s2 = 0.f;
      for (int i = 0; i < 8; ++i) s2 += red[row][i];
      rowinv[row] = 1.0f / s2;
    }
    __syncthreads();
  }

  // ---- O = P V  (each wave handles a 512-key chunk, reduce at the end) ----
  v8f oacc[4];
  for (int nt = 0; nt < 4; ++nt)
    for (int r = 0; r < 8; ++r) oacc[nt][r] = 0.f;

  for (int ks = 0; ks < 16; ++ks) {
    int kbase = wave * 512 + ks * 32;
    // stage V^T: lane owns key kbase+lane (64 contiguous bf16 from global)
    {
      int key = kbase + lane;
      const u32x4* pv = (const u32x4*)(Vp + (rowbase + key) * kDM + h * 64);
      u32x4 tmp[8];
      for (int i = 0; i < 8; ++i) tmp[i] = pv[i];
      const unsigned short* th = (const unsigned short*)tmp;
      for (int d = 0; d < 64; ++d) vt[wave][d][lane] = th[d];
    }
    // per-wave LDS ops are in-order -> reads below see the writes above
    FragBF pa;
    {
      float inv = rowinv[l16];
      for (int e = 0; e < 16; ++e) {
        int kl = (e >> 3) * 16 + half * 8 + (e & 7);
        pa.h[e] = f2bf(sc[l16][kbase + kl] * inv);
      }
    }
    for (int nt = 0; nt < 4; ++nt) {
      FragBF vb;
      vb.q[0] = *(const u32x4*)&vt[wave][nt * 16 + l16][half * 8];
      vb.q[1] = *(const u32x4*)&vt[wave][nt * 16 + l16][16 + half * 8];
      oacc[nt] = __builtin_amdgcn_wmma_f32_16x16x32_bf16(
          false, pa.v, false, vb.v, (short)0, oacc[nt], false, false);
    }
  }

  for (int nt = 0; nt < 4; ++nt)
    for (int r = 0; r < 8; ++r)
      outw[wave][r + 8 * half][nt * 16 + l16] = oacc[nt][r];
  __syncthreads();

  for (int i = t; i < 16 * 64; i += 128) {
    int row = i >> 6, col = i & 63;
    float v = outw[0][row][col] + outw[1][row][col] +
              outw[2][row][col] + outw[3][row][col];
    Op[(rowbase + q0 + row) * kDM + h * 64 + col] = f2bf(v);
  }
}

// ---------------------------------------------------------------------------
extern "C" void kernel_launch(void* const* d_in, const int* in_sizes, int n_in,
                              void* d_out, int out_size, void* d_ws, size_t ws_size,
                              hipStream_t stream) {
  (void)in_sizes; (void)n_in; (void)out_size; (void)ws_size;
  const float* q    = (const float*)d_in[0];
  const float* k    = (const float*)d_in[1];
  const float* v    = (const float*)d_in[2];
  const int*   mask = (const int*)d_in[3];
  const float* w_q  = (const float*)d_in[4];
  const float* b_q  = (const float*)d_in[5];
  const float* w_k  = (const float*)d_in[6];
  const float* b_k  = (const float*)d_in[7];
  const float* w_v  = (const float*)d_in[8];
  const float* b_v  = (const float*)d_in[9];
  const float* w_o  = (const float*)d_in[10];
  const float* b_o  = (const float*)d_in[11];

  const size_t elems = (size_t)4 * kS * kDM;  // 8192*1024
  unsigned short* Qp = (unsigned short*)d_ws;
  unsigned short* Kp = Qp + elems;
  unsigned short* Vp = Kp + elems;
  unsigned short* Ap = Vp + elems;

  dim3 gblk(4 * kS / TM, kDM / TN);  // 64 x 16
  dim3 blk(256);
  gemm_f32_bf16<<<gblk, blk, 0, stream>>>(q, w_q, b_q, Qp);
  gemm_f32_bf16<<<gblk, blk, 0, stream>>>(k, w_k, b_k, Kp);
  gemm_f32_bf16<<<gblk, blk, 0, stream>>>(v, w_v, b_v, Vp);

  dim3 ablk(kS / 16, 16, 4);  // 128 q-tiles x 16 heads x 4 batch
  dim3 ablk_t(128);
  attn_kernel<<<ablk, ablk_t, 0, stream>>>(Qp, Kp, Vp, mask, Ap);

  gemm_bf16_f32<<<gblk, blk, 0, stream>>>(Ap, w_o, b_o, (float*)d_out);
}
